// SNRM_43628277793538
// MI455X (gfx1250) — matrix operand order repaired and checked by reading, not compiled, not run on hardware
//
#include <hip/hip_runtime.h>

typedef __attribute__((ext_vector_type(16))) __bf16 v16bf;
typedef __attribute__((ext_vector_type(8)))  __bf16 v8bf;
typedef __attribute__((ext_vector_type(8)))  float  v8f;

#define BB    64
#define SS    1024
#define EE    300
#define EPAD  320
#define NGRAM 5
#define WN    1020            /* S - (N-1) */
#define K1    1600            /* NGRAM * EPAD, = 50*32 */
#define H0D   512
#define H1D   256
#define H2D   128
#define MTOT  (BB*WN)         /* 65280 = 1020 * 64 */

static __device__ __forceinline__ unsigned short f2bf(float f) {
    unsigned u = __builtin_bit_cast(unsigned, f);
    u += 0x7FFFu + ((u >> 16) & 1u);          // round-to-nearest-even
    return (unsigned short)(u >> 16);
}

// ---------------- embedding gather + cast to padded bf16 ----------------
__global__ void embed_kernel(const int* __restrict__ x, const float* __restrict__ table,
                             unsigned short* __restrict__ ebf) {
    const int t = blockIdx.x;          // b*S + s, one token per block
    const int e = threadIdx.x;         // 0..319
    const int tok = x[t];
    float v = (e < EE) ? table[(long)tok * EE + e] : 0.f;
    ebf[(long)t * EPAD + e] = f2bf(v);
}

// ------------- pack K x N f32 weights into WMMA B-fragment order -------------
// fragment (kt, nt): 32 lanes x 16 halves; lane<16: col=lane, K=kt*32+j (j=0..15)
//                    lane>=16: col=lane-16, K=kt*32+16+j
__global__ void pack_w_kernel(const float* __restrict__ W, unsigned short* __restrict__ Wp, int N) {
    const int frag = blockIdx.x;
    const int ntc = N >> 4;
    const int kt = frag / ntc, nt = frag % ntc;
    const int t = threadIdx.x;                 // 0..511
    const int lane = t >> 4, j = t & 15;
    const int k = kt * 32 + ((lane >= 16) ? 16 : 0) + j;
    const int n = nt * 16 + (lane & 15);
    Wp[(long)frag * 512 + t] = f2bf(W[(long)k * N + n]);
}

// same, but for W_slide [1500,512] expanded to padded K=1600 with zero rows
__global__ void pack_wslide_kernel(const float* __restrict__ W, unsigned short* __restrict__ Wp) {
    const int N = H0D;
    const int frag = blockIdx.x;               // kt*32 + nt  (50 x 32 frags)
    const int kt = frag >> 5, nt = frag & 31;
    const int t = threadIdx.x;
    const int lane = t >> 4, j = t & 15;
    const int kp = kt * 32 + ((lane >= 16) ? 16 : 0) + j;   // padded k in [0,1600)
    const int n = nt * 16 + (lane & 15);
    const int tok = kp / EPAD, e = kp % EPAD;
    float v = (e < EE) ? W[((long)tok * EE + e) * N + n] : 0.f;
    Wp[(long)frag * 512 + t] = f2bf(v);
}

// ---------------- WMMA GEMM: C[M,N] = A[M,K] * W[K,N] + bias ----------------
// block tile 64(M) x 128(N), 256 threads = 8 waves; wave -> 16x64 sub-tile.
// K loop manually unrolled x2 with ping-pong fragment sets (a0/b0 <-> a1/b1):
// loads for one set issue while WMMAs consume the other, and the alternation
// lets the register allocator rotate buffers without copy instructions.
template<int KDIM, int NDIM, bool WINDOWED, bool RELU, bool OUTF32>
__global__ __launch_bounds__(256) void gemm_kernel(
    const unsigned short* __restrict__ Abase,  // WINDOWED: padded emb stream; else row-major [M,KDIM]
    const unsigned short* __restrict__ Wp,     // fragment-packed weights
    const float* __restrict__ bias,
    void* __restrict__ Out)
{
    const int lane  = threadIdx.x & 31;
    const int wave  = threadIdx.x >> 5;
    const int msub  = wave >> 1;                           // 0..3
    const int nhalf = wave & 1;                            // 0..1
    const int m0     = blockIdx.x * 64 + msub * 16;
    const int ntile0 = blockIdx.y * 8 + nhalf * 4;
    const int NT = NDIM >> 4;
    const int KT = KDIM / 32;                              // 50 / 16 / 8: even, >= 2

    // A row handled by this lane (A layout: lane L and L+16 both hold row M=L&15)
    const int row = m0 + (lane & 15);
    long rowbase;
    if (WINDOWED) {
        const int b = row / WN;
        const int w = row - b * WN;
        rowbase = ((long)b * SS + w) * EPAD;               // window = contiguous 1600 elems
    } else {
        rowbase = (long)row * KDIM;
    }
    const int koffA = (lane >= 16) ? 16 : 0;               // lanes>=16 hold K 16..31
    const unsigned short* aptr = Abase + rowbase + koffA;
    const unsigned short* bp0  = Wp + ((long)ntile0 * 32 + lane) * 16;
    const long bstride = (long)NT * 512;                   // elements per K-step of fragments

    auto loadA = [&](int kt) -> v16bf {
        const unsigned short* p = aptr + kt * 32;
        v8bf lo = *(const v8bf*)p;                         // K chunk [koff, koff+8)
        v8bf hi = *(const v8bf*)(p + 8);                   // K chunk [koff+8, koff+16)
        return __builtin_shufflevector(lo, hi,
                    0,1,2,3,4,5,6,7,8,9,10,11,12,13,14,15);
    };

    v8f acc[4] = {v8f{}, v8f{}, v8f{}, v8f{}};
    v16bf a0, a1, b0[4], b1[4];

    // prologue: stage kt=0 into set0
    a0 = loadA(0);
    #pragma unroll
    for (int j = 0; j < 4; ++j) b0[j] = *(const v16bf*)(bp0 + (long)j * 512);

    for (int kt = 0; kt + 2 < KT; kt += 2) {
        // stage kt+1 into set1
        a1 = loadA(kt + 1);
        const unsigned short* bpA = bp0 + (long)(kt + 1) * bstride;
        #pragma unroll
        for (int j = 0; j < 4; ++j) b1[j] = *(const v16bf*)(bpA + (long)j * 512);
        // compute kt from set0
        #pragma unroll
        for (int j = 0; j < 4; ++j)
            acc[j] = __builtin_amdgcn_wmma_f32_16x16x32_bf16(
                         false, a0, false, b0[j], (short)0, acc[j], false, false);
        // stage kt+2 into set0
        a0 = loadA(kt + 2);
        const unsigned short* bpB = bp0 + (long)(kt + 2) * bstride;
        #pragma unroll
        for (int j = 0; j < 4; ++j) b0[j] = *(const v16bf*)(bpB + (long)j * 512);
        // compute kt+1 from set1
        #pragma unroll
        for (int j = 0; j < 4; ++j)
            acc[j] = __builtin_amdgcn_wmma_f32_16x16x32_bf16(
                         false, a1, false, b1[j], (short)0, acc[j], false, false);
    }

    // tail: set0 holds KT-2; stage KT-1 into set1, then compute both
    a1 = loadA(KT - 1);
    const unsigned short* bpT = bp0 + (long)(KT - 1) * bstride;
    #pragma unroll
    for (int j = 0; j < 4; ++j) b1[j] = *(const v16bf*)(bpT + (long)j * 512);
    #pragma unroll
    for (int j = 0; j < 4; ++j)
        acc[j] = __builtin_amdgcn_wmma_f32_16x16x32_bf16(
                     false, a0, false, b0[j], (short)0, acc[j], false, false);
    #pragma unroll
    for (int j = 0; j < 4; ++j)
        acc[j] = __builtin_amdgcn_wmma_f32_16x16x32_bf16(
                     false, a1, false, b1[j], (short)0, acc[j], false, false);

    // epilogue: C layout -> lane col = lane&15, VGPR r -> row m0 + r + (lane>=16?8:0)
    const int rbase = (lane >= 16) ? 8 : 0;
    #pragma unroll
    for (int j = 0; j < 4; ++j) {
        const int n = (ntile0 + j) * 16 + (lane & 15);
        const float bv = bias[n];
        #pragma unroll
        for (int r = 0; r < 8; ++r) {
            float v = acc[j][r] + bv;
            if (RELU) v = v > 0.f ? v : 0.f;
            const long rr = (long)(m0 + rbase + r) * NDIM + n;
            if (OUTF32) ((float*)Out)[rr] = v;
            else        ((unsigned short*)Out)[rr] = f2bf(v);
        }
    }
}

// ---------------- ragged-mean pooling over valid windows ----------------
// 512 threads: 4 window-quarters x 128 channels, LDS tree reduce.
__global__ __launch_bounds__(512) void pool_kernel(
    const float* __restrict__ H2p, const int* __restrict__ lengths,
    float* __restrict__ out) {
    __shared__ float red[512];
    const int b = blockIdx.x;
    const int t = threadIdx.x;
    const int h = t & (H2D - 1);
    const int q = t >> 7;                      // 0..3
    const int valid = lengths[b] - (NGRAM - 1);
    const float* p = H2p + (long)b * WN * H2D + h;
    float s = 0.f;
    for (int w = q; w < valid; w += 4) s += p[(long)w * H2D];
    red[t] = s;
    __syncthreads();
    if (q == 0) {
        float tot = red[h] + red[h + 128] + red[h + 256] + red[h + 384];
        out[b * H2D + h] = tot / (float)valid;
    }
}

extern "C" void kernel_launch(void* const* d_in, const int* in_sizes, int n_in,
                              void* d_out, int out_size, void* d_ws, size_t ws_size,
                              hipStream_t stream) {
    (void)in_sizes; (void)n_in; (void)out_size; (void)ws_size;
    const int*   x      = (const int*)d_in[0];
    const int*   len    = (const int*)d_in[1];
    const float* table  = (const float*)d_in[2];
    const float* Wsl    = (const float*)d_in[3];
    const float* bsl    = (const float*)d_in[4];
    const float* W1     = (const float*)d_in[5];
    const float* b1v    = (const float*)d_in[6];
    const float* W2     = (const float*)d_in[7];
    const float* b2v    = (const float*)d_in[8];

    char* ws = (char*)d_ws;                      // 256B-aligned region offsets
    unsigned short* Ebf = (unsigned short*)(ws);                       // 41,943,040 B
    unsigned short* Wp1 = (unsigned short*)(ws + 41943040UL);          //  1,638,400 B
    unsigned short* Wp2 = (unsigned short*)(ws + 43581440UL);          //    262,144 B
    unsigned short* Wp3 = (unsigned short*)(ws + 43843584UL);          //     65,536 B
    unsigned short* H0b = (unsigned short*)(ws + 43909120UL);          // 66,846,720 B
    unsigned short* H1b = (unsigned short*)(ws + 110755840UL);         // 33,423,360 B
    float*          H2b = (float*)        (ws + 144179200UL);          // 33,423,360 B

    embed_kernel<<<BB * SS, EPAD, 0, stream>>>(x, table, Ebf);
    pack_wslide_kernel<<<(K1/32)*(H0D/16), 512, 0, stream>>>(Wsl, Wp1);
    pack_w_kernel<<<(H0D/32)*(H1D/16), 512, 0, stream>>>(W1, Wp2, H1D);
    pack_w_kernel<<<(H1D/32)*(H2D/16), 512, 0, stream>>>(W2, Wp3, H2D);

    gemm_kernel<K1,  H0D, true,  false, false>
        <<<dim3(MTOT/64, H0D/128), 256, 0, stream>>>(Ebf, Wp1, bsl, H0b);
    gemm_kernel<H0D, H1D, false, true,  false>
        <<<dim3(MTOT/64, H1D/128), 256, 0, stream>>>(H0b, Wp2, b1v, H1b);
    gemm_kernel<H1D, H2D, false, true,  true>
        <<<dim3(MTOT/64, H2D/128), 256, 0, stream>>>(H1b, Wp3, b2v, H2b);

    pool_kernel<<<BB, 512, 0, stream>>>(H2b, len, (float*)d_out);
}
